// AttentionBlock_72499047957007
// MI455X (gfx1250) — matrix operand order repaired and checked by reading, not compiled, not run on hardware
//
#include <hip/hip_runtime.h>
#include <hip/hip_bf16.h>
#include <math.h>

#define BB 128
#define NNTOK 256
#define CCH 512
#define HH 16
#define DD 32
#define SCALE_Q 0.17677669529663689f

typedef _Float16 v16h __attribute__((ext_vector_type(16)));
typedef _Float16 h8   __attribute__((ext_vector_type(8)));
typedef float    v8f  __attribute__((ext_vector_type(8)));

#define WMMA_F16(a, b, c) \
  __builtin_amdgcn_wmma_f32_16x16x32_f16(false, (a), false, (b), (short)0, (c), false, false)

// Load a 16x32 f16 A-fragment: lane (l,hi) holds row m=l; K range
// 16*hi..16*hi+15 (contiguous 16-byte loads).
__device__ __forceinline__ v16h load_a16(const _Float16* p, int hi) {
  h8 lo = *(const h8*)(p + 16 * hi);
  h8 hv = *(const h8*)(p + 16 * hi + 8);
  v16h a;
#pragma unroll
  for (int t = 0; t < 8; ++t) { a[t] = lo[t]; a[8 + t] = hv[t]; }
  return a;
}

// ---------------- prep kernels ----------------
__global__ __launch_bounds__(256) void k_cvt(const float* __restrict__ src,
                                             _Float16* __restrict__ dst, int n) {
  int i = blockIdx.x * 256 + threadIdx.x;
  if (i < n) dst[i] = (_Float16)src[i];
}

__global__ __launch_bounds__(256) void k_bias(const float* __restrict__ tab,
                                              float* __restrict__ biasM) {
  int idx = blockIdx.x * 256 + threadIdx.x;     // H*N*N = 1048576
  int j = idx & 255, i = (idx >> 8) & 255, h = idx >> 16;
  int ri = i >> 4, ci = i & 15, rj = j >> 4, cj = j & 15;
  int t = (ri - rj + 15) * 31 + (ci - cj + 15);
  biasM[idx] = tab[t * HH + h];
}

__global__ __launch_bounds__(256) void k_mod(float* __restrict__ modM) {
  int idx = blockIdx.x * 256 + threadIdx.x;     // N*N = 65536
  int j = idx & 255, i = idx >> 8;
  double ri = (double)(i >> 4), ci = (double)(i & 15);
  double rj = (double)(j >> 4), cj = (double)(j & 15);
  double d = sqrt((ri - rj) * (ri - rj) + (ci - cj) * (ci - cj));
  double T = 4.0 * 15.0 * sqrt(2.0);
  double f = 2.0 * 3.14159265358979323846 / T;
  double m = exp(cos(f * d)) * 0.5;
  double bound = exp(cos(f * sqrt(229.0))) * 0.5;  // m[0, 47]
  modM[idx] = (float)((m < bound) ? 0.0 : m);
}

// ---------------- QKV GEMM: [32768,512] @ [512,1536]^T ----------------
__global__ __launch_bounds__(256) void k_qkv(const _Float16* __restrict__ xh,
                                             const _Float16* __restrict__ wh,
                                             const float* __restrict__ bvec,
                                             _Float16* __restrict__ qh,
                                             _Float16* __restrict__ kh,
                                             _Float16* __restrict__ vt) {
  const int lane = threadIdx.x & 31, l = lane & 15, hi = lane >> 4;
  const int wave = threadIdx.x >> 5;
  const int job = blockIdx.x * 8 + wave;    // 49152 jobs
  const int mt = job / 24;                  // 0..2047 (16-row tiles)
  const int nt = job % 24;                  // 0..23 (64-col strips)
  v8f z = {};
  v8f acc[4] = {z, z, z, z};
  const _Float16* arow = xh + (size_t)(mt * 16 + l) * CCH;
#pragma unroll 4
  for (int k0 = 0; k0 < CCH; k0 += 32) {
    v16h a = load_a16(arow + k0, hi);
#pragma unroll
    for (int nn = 0; nn < 4; ++nn) {
      int wr = nt * 64 + nn * 16 + l;       // weight row == output column
      v16h b = *(const v16h*)(wh + (size_t)wr * CCH + k0 + 16 * hi);
      acc[nn] = WMMA_F16(a, b, acc[nn]);
    }
  }
#pragma unroll
  for (int nn = 0; nn < 4; ++nn) {
    int gc = nt * 64 + nn * 16 + l;
    float bv = bvec[gc];
    int which = gc >> 9;                    // 0=q, 1=k, 2=v
    int hch = (gc >> 5) & 15;
    int d = gc & 31;
#pragma unroll
    for (int r = 0; r < 8; ++r) {
      int gm = mt * 16 + r + 8 * hi;
      int bw = gm >> 8, tok = gm & 255;
      float val = acc[nn][r] + bv;
      if (which == 0)
        qh[(size_t)(((bw * HH + hch) * NNTOK + tok)) * DD + d] = (_Float16)(val * SCALE_Q);
      else if (which == 1)
        kh[(size_t)(((bw * HH + hch) * NNTOK + tok)) * DD + d] = (_Float16)val;
      else
        vt[(size_t)(((bw * HH + hch) * DD + d)) * NNTOK + tok] = (_Float16)val;  // transposed
    }
  }
}

// ---------------- attention: one block per (b,h) ----------------
__global__ __launch_bounds__(256) void k_attn(const _Float16* __restrict__ qh,
                                              const _Float16* __restrict__ kh,
                                              const _Float16* __restrict__ vt,
                                              const float* __restrict__ biasM,
                                              const float* __restrict__ modM,
                                              _Float16* __restrict__ aoh) {
  __shared__ _Float16 sP[8 * 4096];         // 64 KB: per-wave 16x256 P tile
  const int lane = threadIdx.x & 31, l = lane & 15, hi = lane >> 4;
  const int wave = threadIdx.x >> 5;
  const int b = blockIdx.x >> 4, h = blockIdx.x & 15;
  const _Float16* qb = qh + (size_t)((b * HH + h) * NNTOK) * DD;
  const _Float16* kb = kh + (size_t)((b * HH + h) * NNTOK) * DD;
  const _Float16* vb = vt + (size_t)((b * HH + h) * DD) * NNTOK;
  const float* bb = biasM + (size_t)h * NNTOK * NNTOK;
  _Float16* lds = sP + wave * 4096;

  for (int rb = 0; rb < 2; ++rb) {
    const int i0 = (wave * 2 + rb) * 16;
    v16h aq = load_a16(qb + (i0 + l) * DD, hi);      // q rows i0..i0+15, K=D=32
    v8f s[16];
#pragma unroll
    for (int jt = 0; jt < 16; ++jt) {
      v8f c;
#pragma unroll
      for (int r = 0; r < 8; ++r)                    // bias preloaded as C operand
        c[r] = bb[(size_t)(i0 + r + 8 * hi) * NNTOK + jt * 16 + l];
      v16h bk = *(const v16h*)(kb + (jt * 16 + l) * DD + 16 * hi);
      s[jt] = WMMA_F16(aq, bk, c);
    }
    // row softmax (rows live in 16-lane halves; xor masks 1..8 stay in-half)
    float inv[8];
#pragma unroll
    for (int r = 0; r < 8; ++r) {
      float mx = -3.4e38f;
#pragma unroll
      for (int jt = 0; jt < 16; ++jt) mx = fmaxf(mx, s[jt][r]);
      for (int msk = 1; msk < 16; msk <<= 1) mx = fmaxf(mx, __shfl_xor(mx, msk, 32));
      float sum = 0.f;
#pragma unroll
      for (int jt = 0; jt < 16; ++jt) {
        float e = expf(s[jt][r] - mx);
        s[jt][r] = e;
        sum += e;
      }
      for (int msk = 1; msk < 16; msk <<= 1) sum += __shfl_xor(sum, msk, 32);
      inv[r] = 1.0f / sum;
    }
    // normalize, apply modulation, park P in LDS as f16 row-major
#pragma unroll
    for (int jt = 0; jt < 16; ++jt) {
#pragma unroll
      for (int r = 0; r < 8; ++r) {
        int row = r + 8 * hi;
        float p = s[jt][r] * inv[r] * modM[(size_t)(i0 + row) * NNTOK + jt * 16 + l];
        lds[row * 256 + jt * 16 + l] = (_Float16)p;
      }
    }
    // out[16x32] = P[16x256] @ V[256x32]; A from LDS, B from transposed V
    v8f o0 = {}, o1 = {};
#pragma unroll
    for (int kt = 0; kt < 8; ++kt) {
      v16h ap = load_a16(lds + l * 256 + kt * 32, hi);
      v16h b0 = *(const v16h*)(vb + (0 + l) * NNTOK + kt * 32 + 16 * hi);
      o0 = WMMA_F16(ap, b0, o0);
      v16h b1 = *(const v16h*)(vb + (16 + l) * NNTOK + kt * 32 + 16 * hi);
      o1 = WMMA_F16(ap, b1, o1);
    }
#pragma unroll
    for (int r = 0; r < 8; ++r) {
      int i = i0 + r + 8 * hi;
      _Float16* op = aoh + (size_t)(b * NNTOK + i) * CCH + h * DD;
      op[l] = (_Float16)o0[r];
      op[16 + l] = (_Float16)o1[r];
    }
  }
}

// ---------------- output proj: [32768,512] @ [512,512]^T + b ----------------
__global__ __launch_bounds__(256) void k_proj(const _Float16* __restrict__ aoh,
                                              const _Float16* __restrict__ wh,
                                              const float* __restrict__ bvec,
                                              float* __restrict__ out) {
  const int lane = threadIdx.x & 31, l = lane & 15, hi = lane >> 4;
  const int wave = threadIdx.x >> 5;
  const int job = blockIdx.x * 8 + wave;    // 16384 jobs
  const int mt = job >> 3;
  const int nt = job & 7;
  v8f z = {};
  v8f acc[4] = {z, z, z, z};
  const _Float16* arow = aoh + (size_t)(mt * 16 + l) * CCH;
#pragma unroll 4
  for (int k0 = 0; k0 < CCH; k0 += 32) {
    v16h a = load_a16(arow + k0, hi);
#pragma unroll
    for (int nn = 0; nn < 4; ++nn) {
      int wr = nt * 64 + nn * 16 + l;
      v16h b = *(const v16h*)(wh + (size_t)wr * CCH + k0 + 16 * hi);
      acc[nn] = WMMA_F16(a, b, acc[nn]);
    }
  }
#pragma unroll
  for (int nn = 0; nn < 4; ++nn) {
    int gc = nt * 64 + nn * 16 + l;
    float bv = bvec[gc];
#pragma unroll
    for (int r = 0; r < 8; ++r) {
      int gm = mt * 16 + r + 8 * hi;
      out[(size_t)gm * CCH + gc] = acc[nn][r] + bv;
    }
  }
}

extern "C" void kernel_launch(void* const* d_in, const int* in_sizes, int n_in,
                              void* d_out, int out_size, void* d_ws, size_t ws_size,
                              hipStream_t stream) {
  (void)in_sizes; (void)n_in; (void)out_size; (void)ws_size;
  const float* x      = (const float*)d_in[0];
  const float* qkv_w  = (const float*)d_in[1];
  const float* qkv_b  = (const float*)d_in[2];
  const float* proj_w = (const float*)d_in[3];
  const float* proj_b = (const float*)d_in[4];
  const float* rpb    = (const float*)d_in[5];
  float* out = (float*)d_out;

  char* ws = (char*)d_ws;
  size_t off = 0;
  auto take = [&](size_t bytes) { char* p = ws + off; off += (bytes + 255) & ~(size_t)255; return p; };
  _Float16* xh    = (_Float16*)take((size_t)BB * NNTOK * CCH * 2);        // 32 MB
  _Float16* wqkv  = (_Float16*)take((size_t)3 * CCH * CCH * 2);
  _Float16* wproj = (_Float16*)take((size_t)CCH * CCH * 2);
  _Float16* qh    = (_Float16*)take((size_t)BB * HH * NNTOK * DD * 2);
  _Float16* kh    = (_Float16*)take((size_t)BB * HH * NNTOK * DD * 2);
  _Float16* vt    = (_Float16*)take((size_t)BB * HH * DD * NNTOK * 2);
  _Float16* aoh   = (_Float16*)take((size_t)BB * NNTOK * CCH * 2);
  float* biasM    = (float*)take((size_t)HH * NNTOK * NNTOK * 4);         // 4 MB
  float* modM     = (float*)take((size_t)NNTOK * NNTOK * 4);

  const int nx = BB * NNTOK * CCH;          // 16777216
  const int nwq = 3 * CCH * CCH;            // 786432
  const int nwp = CCH * CCH;                // 262144

  k_cvt<<<(nx + 255) / 256, 256, 0, stream>>>(x, xh, nx);
  k_cvt<<<(nwq + 255) / 256, 256, 0, stream>>>(qkv_w, wqkv, nwq);
  k_cvt<<<(nwp + 255) / 256, 256, 0, stream>>>(proj_w, wproj, nwp);
  k_bias<<<(HH * NNTOK * NNTOK) / 256, 256, 0, stream>>>(rpb, biasM);
  k_mod<<<(NNTOK * NNTOK) / 256, 256, 0, stream>>>(modM);

  // QKV: 2048 M-tiles * 24 N-strips = 49152 waves -> 6144 blocks of 8 waves
  k_qkv<<<6144, 256, 0, stream>>>(xh, wqkv, qkv_b, qh, kh, vt);
  // Attention: one block per (b,h)
  k_attn<<<BB * HH, 256, 0, stream>>>(qh, kh, vt, biasM, modM, aoh);
  // Proj: 2048 M-tiles * 8 N-strips = 16384 waves -> 2048 blocks
  k_proj<<<2048, 256, 0, stream>>>(aoh, wproj, proj_b, out);
}
